// QLSTM_65481071395545
// MI455X (gfx1250) — matrix operand order repaired and checked by reading, not compiled, not run on hardware
//
#include <hip/hip_runtime.h>

typedef __attribute__((ext_vector_type(16))) _Float16 v16h;
typedef __attribute__((ext_vector_type(8)))  float    v8f;

#define T_STEPS 48
#define BATCH   1024
#define IN_DIM  512
#define NQ      6
#define FAN_IN  518     // 512 + 6
#define NCOL    24      // 4 gates * 6 qubits
#define NCOLP   32      // padded to two 16-wide WMMA N tiles

// workspace layout
#define WS_WC_OFF   0        // 32*512 f16 = 32768 B  (padded weight matrix, K-major rows)
#define WS_BC_OFF   32768    // 32 f32               (bias + theta, padded)
#define WS_PRE_OFF  36864    // T*B*24 f32 = 4.72 MB (input-projection pre-activations)

// ---------------------------------------------------------------------------
// Kernel 1: pack weights (x-columns only) into padded f16 [32][512], fold
// bias+theta into bcat[32].
// ---------------------------------------------------------------------------
__global__ __launch_bounds__(256) void qlstm_prep(
    const float* __restrict__ Wf, const float* __restrict__ Wi,
    const float* __restrict__ Wg, const float* __restrict__ Wo,
    const float* __restrict__ bfp, const float* __restrict__ bip,
    const float* __restrict__ bgp, const float* __restrict__ bop,
    const float* __restrict__ thf, const float* __restrict__ thi,
    const float* __restrict__ thg, const float* __restrict__ tho,
    _Float16* __restrict__ Wc, float* __restrict__ bcat)
{
    const int idx = blockIdx.x * 256 + threadIdx.x;   // 0 .. 16383
    const int n = idx >> 9;                           // 0 .. 31 (output column)
    const int k = idx & 511;                          // 0 .. 511 (input dim)
    float v = 0.0f;
    if (n < NCOL) {
        const int g = n / NQ, q = n % NQ;
        const float* W = (g == 0) ? Wf : (g == 1) ? Wi : (g == 2) ? Wg : Wo;
        v = W[q * FAN_IN + k];
    }
    Wc[(size_t)n * IN_DIM + k] = (_Float16)v;
    if (idx < NCOLP) {
        float bb = 0.0f;
        if (idx < NCOL) {
            const int g = idx / NQ, q = idx % NQ;
            const float* bp = (g == 0) ? bfp : (g == 1) ? bip : (g == 2) ? bgp : bop;
            const float* tp = (g == 0) ? thf : (g == 1) ? thi : (g == 2) ? thg : tho;
            bb = bp[q] + tp[q];
        }
        bcat[idx] = bb;
    }
}

// ---------------------------------------------------------------------------
// Kernel 2: pre[T*B][24] = X[T*B][512] @ Wc^T + bcat  via v_wmma_f32_16x16x32_f16.
// One wave per 16-row M tile; each wave computes BOTH 16-wide N tiles so the
// A fragment (the HBM-bound X stream) is fetched exactly once. 3072 wave
// tasks, 8 waves/block -> 384 blocks. f32 -> f16 conversion is in-register.
// ---------------------------------------------------------------------------
__global__ __launch_bounds__(256) void qlstm_gemm(
    const float* __restrict__ X, const _Float16* __restrict__ Wc,
    const float* __restrict__ bcat, float* __restrict__ pre)
{
    const int lane  = threadIdx.x & 31;
    const int wave  = threadIdx.x >> 5;
    const int mtile = blockIdx.x * 8 + wave;   // 0 .. 3071
    const int m16   = lane & 15;
    const int half  = lane >> 4;               // lane half selects K sub-block

    const float*    xrow  = X  + (size_t)(mtile * 16 + m16) * IN_DIM;  // A row
    const _Float16* wrow0 = Wc + (size_t)m16 * IN_DIM;                 // B cols 0..15
    const _Float16* wrow1 = Wc + (size_t)(16 + m16) * IN_DIM;          // B cols 16..31 (rows 24..31 zero)

    v8f acc0 = {};
    v8f acc1 = {};
    #pragma unroll 4
    for (int ks = 0; ks < 16; ++ks) {
        const int k0 = ks * 32;
        // A fragment (16x32 f16): lanes<16 hold K k0+0..15,
        //                         lanes>=16 hold K k0+16..31
        const float4* pa = (const float4*)(xrow + k0 + half * 16);
        const float4* pb = (const float4*)(xrow + k0 + half * 16 + 8);
        const float4 a0 = pa[0], a1 = pa[1];
        const float4 a2 = pb[0], a3 = pb[1];
        v16h A;
        A[0]  = (_Float16)a0.x; A[1]  = (_Float16)a0.y; A[2]  = (_Float16)a0.z; A[3]  = (_Float16)a0.w;
        A[4]  = (_Float16)a1.x; A[5]  = (_Float16)a1.y; A[6]  = (_Float16)a1.z; A[7]  = (_Float16)a1.w;
        A[8]  = (_Float16)a2.x; A[9]  = (_Float16)a2.y; A[10] = (_Float16)a2.z; A[11] = (_Float16)a2.w;
        A[12] = (_Float16)a3.x; A[13] = (_Float16)a3.y; A[14] = (_Float16)a3.z; A[15] = (_Float16)a3.w;
        // B fragments (32x16 f16): lanes<16 hold K k0+0..15, lanes>=16 hold K k0+16..31
        const v16h B0 = *(const v16h*)(wrow0 + k0 + half * 16);
        const v16h B1 = *(const v16h*)(wrow1 + k0 + half * 16);
        acc0 = __builtin_amdgcn_wmma_f32_16x16x32_f16(
            false, A, false, B0, (short)0, acc0, false, false);
        acc1 = __builtin_amdgcn_wmma_f32_16x16x32_f16(
            false, A, false, B1, (short)0, acc1, false, false);
    }

    // C/D layout: VGPR i -> row (i + 8*half), lane%16 -> column
    const int mbase = mtile * 16 + half * 8;
    {
        const float bias = bcat[m16];                 // cols 0..15, always valid
        #pragma unroll
        for (int i = 0; i < 8; ++i)
            pre[(size_t)(mbase + i) * NCOL + m16] = acc0[i] + bias;
    }
    const int ncol1 = 16 + m16;
    if (ncol1 < NCOL) {                               // cols 16..23 valid, 24..31 padding
        const float bias = bcat[ncol1];
        #pragma unroll
        for (int i = 0; i < 8; ++i)
            pre[(size_t)(mbase + i) * NCOL + ncol1] = acc1[i] + bias;
    }
}

// ---------------------------------------------------------------------------
// Kernel 3: sequential scan over T. One thread per batch element.
// Quantum layer in closed form: z products of cos(angle) per the CNOT-ring
// bit map; then standard LSTM cell update.
// ---------------------------------------------------------------------------
__global__ __launch_bounds__(256) void qlstm_recur(
    const float* __restrict__ pre,
    const float* __restrict__ Wf, const float* __restrict__ Wi,
    const float* __restrict__ Wg, const float* __restrict__ Wo,
    float* __restrict__ out)
{
    const int b = blockIdx.x * 256 + threadIdx.x;
    if (b >= BATCH) return;

    // recurrent 6x6 blocks (columns 512..517 of each W)
    float Wh[4][NQ][NQ];
    #pragma unroll
    for (int q = 0; q < NQ; ++q) {
        #pragma unroll
        for (int j = 0; j < NQ; ++j) {
            Wh[0][q][j] = Wf[q * FAN_IN + IN_DIM + j];
            Wh[1][q][j] = Wi[q * FAN_IN + IN_DIM + j];
            Wh[2][q][j] = Wg[q * FAN_IN + IN_DIM + j];
            Wh[3][q][j] = Wo[q * FAN_IN + IN_DIM + j];
        }
    }

    float hx[NQ] = {0.f, 0.f, 0.f, 0.f, 0.f, 0.f};
    float cx[NQ] = {0.f, 0.f, 0.f, 0.f, 0.f, 0.f};

    for (int t = 0; t < T_STEPS; ++t) {
        const float* p = pre + ((size_t)t * BATCH + b) * NCOL;
        float zv[4][NQ];
        #pragma unroll
        for (int g = 0; g < 4; ++g) {
            float ca[NQ];
            #pragma unroll
            for (int q = 0; q < NQ; ++q) {
                float a = p[g * NQ + q];
                #pragma unroll
                for (int j = 0; j < NQ; ++j) a += Wh[g][q][j] * hx[j];
                ca[q] = cosf(a);
            }
            // CNOT-ring measurement: y1=x0^x1, ..., y5=x0^..^x5, y0=x1^..^x5
            zv[g][1] = ca[0] * ca[1];
            zv[g][2] = zv[g][1] * ca[2];
            zv[g][3] = zv[g][2] * ca[3];
            zv[g][4] = zv[g][3] * ca[4];
            zv[g][5] = zv[g][4] * ca[5];
            zv[g][0] = ca[1] * ca[2] * ca[3] * ca[4] * ca[5];
        }
        #pragma unroll
        for (int q = 0; q < NQ; ++q) {
            const float fg = 1.0f / (1.0f + expf(-zv[0][q]));
            const float ig = 1.0f / (1.0f + expf(-zv[1][q]));
            const float gg = tanhf(zv[2][q]);
            const float og = 1.0f / (1.0f + expf(-zv[3][q]));
            cx[q] = fg * cx[q] + ig * gg;
            hx[q] = og * tanhf(cx[q]);
            out[((size_t)t * BATCH + b) * NQ + q] = hx[q];
        }
    }
    // trailing hx, cx (reference returns outputs, hx, cx concatenated flat)
    #pragma unroll
    for (int q = 0; q < NQ; ++q) {
        out[(size_t)T_STEPS * BATCH * NQ + (size_t)b * NQ + q] = hx[q];
        out[(size_t)T_STEPS * BATCH * NQ + (size_t)BATCH * NQ + (size_t)b * NQ + q] = cx[q];
    }
}

extern "C" void kernel_launch(void* const* d_in, const int* in_sizes, int n_in,
                              void* d_out, int out_size, void* d_ws, size_t ws_size,
                              hipStream_t stream) {
    const float* X   = (const float*)d_in[0];
    const float* Wf  = (const float*)d_in[1];
    const float* bf_ = (const float*)d_in[2];
    const float* Wi  = (const float*)d_in[3];
    const float* bi_ = (const float*)d_in[4];
    const float* Wg  = (const float*)d_in[5];
    const float* bg_ = (const float*)d_in[6];
    const float* Wo  = (const float*)d_in[7];
    const float* bo_ = (const float*)d_in[8];
    const float* thf = (const float*)d_in[9];
    const float* thi = (const float*)d_in[10];
    const float* thg = (const float*)d_in[11];
    const float* tho = (const float*)d_in[12];
    float* out = (float*)d_out;

    _Float16* Wc   = (_Float16*)((char*)d_ws + WS_WC_OFF);
    float*    bcat = (float*)   ((char*)d_ws + WS_BC_OFF);
    float*    pre  = (float*)   ((char*)d_ws + WS_PRE_OFF);

    (void)in_sizes; (void)n_in; (void)out_size; (void)ws_size;

    qlstm_prep <<<64,  256, 0, stream>>>(Wf, Wi, Wg, Wo, bf_, bi_, bg_, bo_,
                                         thf, thi, thg, tho, Wc, bcat);
    qlstm_gemm <<<384, 256, 0, stream>>>(X, Wc, bcat, pre);   // 3072 wave tasks
    qlstm_recur<<<4,   256, 0, stream>>>(pre, Wf, Wi, Wg, Wo, out);
}
